// MeshDeformationNet_69655779607185
// MI455X (gfx1250) — matrix operand order repaired and checked by reading, not compiled, not run on hardware
//
#include <hip/hip_runtime.h>
#include <hip/hip_bf16.h>
#include <math.h>

#define NNODES 100000
#define NEDGES 1600000

typedef __attribute__((ext_vector_type(16))) __bf16 v16bf;
typedef __attribute__((ext_vector_type(8)))  __bf16 v8bf;
typedef __attribute__((ext_vector_type(8)))  float  v8f;

__device__ __forceinline__ unsigned short f32_to_bf16_rne(float f) {
  unsigned int u = __float_as_uint(f);
  u += 0x7fffu + ((u >> 16) & 1u);
  return (unsigned short)(u >> 16);
}

__device__ __forceinline__ void atomicMaxF(float* addr, float val) {
  if (val >= 0.f) atomicMax((int*)addr, __float_as_int(val));
  else            atomicMin((unsigned int*)addr, __float_as_uint(val));
}

// ---------- plain dense: fc1 (K=23) and output head (Cout=3) ----------
__global__ void dense_naive_kernel(const float* __restrict__ X, const float* __restrict__ W,
                                   const float* __restrict__ b, float* __restrict__ Y,
                                   int N, int K, int Cout) {
  long long i = (long long)blockIdx.x * blockDim.x + threadIdx.x;
  if (i >= (long long)N * Cout) return;
  int n = (int)(i / Cout), co = (int)(i - (long long)n * Cout);
  const float* xr = X + (long long)n * K;
  const float* wr = W + (long long)co * K;
  float s = b[co];
  for (int k = 0; k < K; ++k) s = fmaf(xr[k], wr[k], s);
  Y[i] = s;
}

// ---------- f32 -> bf16 convert (weights / skip activations) ----------
__global__ void cvt_bf16_kernel(const float* __restrict__ in,
                                unsigned short* __restrict__ out, long long n) {
  long long i = (long long)blockIdx.x * blockDim.x + threadIdx.x;
  if (i < n) out[i] = f32_to_bf16_rne(in[i]);
}

// ---------- WMMA GEMM: C[M,Cout] = A[M,K](bf16) * W[Cout,K](bf16)^T + bias ----------
// 256-thread block = 8 waves, each wave owns one 16-row M tile and computes a
// 16 x (NT*16) output strip. The block's weight strip [NT*16, K] is staged once
// into LDS (cooperative global->LDS copy, then ds_load fragments reused by all
// 8 waves). NT=4 when Cout%64==0, NT=2 for Cout==32. K, Cout multiples of 32.
template <int NT>
__global__ __launch_bounds__(256)
void wmma_gemm_kernel(const unsigned short* __restrict__ A,
                      const unsigned short* __restrict__ W,
                      const float* __restrict__ bias,
                      float* __restrict__ C, int M, int K, int Cout) {
  __shared__ __align__(32) unsigned short ldsW[16 * NT * 256];  // up to 32 KB
  const int tid   = threadIdx.x;
  const int nBase = blockIdx.y * (16 * NT);

  // cooperative load: weight strip rows nBase..nBase+16*NT-1, all K, 16B chunks
  const int k8 = K >> 3;                // 16B chunks per row
  const int chunks = 16 * NT * k8;
  for (int idx = tid; idx < chunks; idx += 256) {
    int row  = idx / k8;
    int col8 = idx - row * k8;
    *(v8bf*)(ldsW + (size_t)row * K + (col8 << 3)) =
        *(const v8bf*)(W + (size_t)(nBase + row) * K + (col8 << 3));
  }
  __syncthreads();

  const int wave = tid >> 5;
  const int lane = tid & 31;
  const int r    = lane & 15;
  const int hi   = lane >> 4;
  const int m0   = (blockIdx.x * 8 + wave) * 16;
  if (m0 >= M) return;                  // wave-uniform: EXEC stays all-ones

  v8f acc[NT];
#pragma unroll
  for (int t = 0; t < NT; ++t) acc[t] = {};

  // A fragment: lane holds row m0+r; lo lanes K {0..7,16..23}, hi lanes {8..15,24..31}
  const unsigned short* aptr  = A + (size_t)(m0 + r) * K + hi * 8;
  // B fragment from LDS: lane holds col (t*16 + r); lo lanes K {0..7,16..23}, hi lanes {8..15,24..31}
  const unsigned short* bbase = ldsW + (size_t)r * K + hi * 8;

  for (int kb = 0; kb < K; kb += 32) {
    v8bf a0 = *(const v8bf*)(aptr + kb);
    v8bf a1 = *(const v8bf*)(aptr + kb + 16);
    v16bf av;
#pragma unroll
    for (int j = 0; j < 8; ++j) { av[j] = a0[j]; av[j + 8] = a1[j]; }
#pragma unroll
    for (int t = 0; t < NT; ++t) {
      v8bf b0 = *(const v8bf*)(bbase + (size_t)t * 16 * K + kb);
      v8bf b1 = *(const v8bf*)(bbase + (size_t)t * 16 * K + kb + 16);
      v16bf bv;
#pragma unroll
      for (int j = 0; j < 8; ++j) { bv[j] = b0[j]; bv[j + 8] = b1[j]; }
      acc[t] = __builtin_amdgcn_wmma_f32_16x16x32_bf16(false, av, false, bv,
                                                       (short)0, acc[t], false, false);
    }
  }
#pragma unroll
  for (int t = 0; t < NT; ++t) {
    const int cn = nBase + t * 16 + r;
    const float bn = bias ? bias[cn] : 0.f;
#pragma unroll
    for (int j = 0; j < 8; ++j)
      C[(size_t)(m0 + hi * 8 + j) * Cout + cn] = acc[t][j] + bn;
  }
}

// ---------- GroupNorm (8 ch/group) + ELU; writes fp32 and/or bf16 ----------
__global__ void gn_elu_kernel(const float* __restrict__ x, const float* __restrict__ w,
                              const float* __restrict__ b, float* __restrict__ outf,
                              unsigned short* __restrict__ outb, int N, int C) {
  const int G = C >> 3;
  long long i = (long long)blockIdx.x * blockDim.x + threadIdx.x;
  if (i >= (long long)N * G) return;
  int n = (int)(i / G), g = (int)(i - (long long)n * G);
  const float* xr = x + (long long)n * C + g * 8;
  float mu = 0.f, m2 = 0.f;
#pragma unroll
  for (int j = 0; j < 8; ++j) { float v = xr[j]; mu += v; m2 += v * v; }
  mu *= 0.125f;
  float var = m2 * 0.125f - mu * mu;
  float rs  = rsqrtf(var + 1e-5f);
  long long o = (long long)n * C + g * 8;
#pragma unroll
  for (int j = 0; j < 8; ++j) {
    float y = (xr[j] - mu) * rs * w[g * 8 + j] + b[g * 8 + j];
    y = y > 0.f ? y : expm1f(y);            // ELU
    if (outf) outf[o + j] = y;
    if (outb) outb[o + j] = f32_to_bf16_rne(y);
  }
}

// ---------- GAT: per-node attention scores ----------
__global__ void gat_scores_kernel(const float* __restrict__ h, const float* __restrict__ as_,
                                  const float* __restrict__ ad_, float* __restrict__ aS,
                                  float* __restrict__ aD, int N, int half) {
  int n = blockIdx.x * blockDim.x + threadIdx.x;
  if (n >= N) return;
  const float* hr = h + (long long)n * half;
  float s = 0.f, d = 0.f;
  for (int c2 = 0; c2 < half; ++c2) { float v = hr[c2]; s = fmaf(v, as_[c2], s); d = fmaf(v, ad_[c2], d); }
  aS[n] = s; aD[n] = d;
}

__global__ void gat_init_kernel(float* __restrict__ amax, float* __restrict__ den, int N) {
  int n = blockIdx.x * blockDim.x + threadIdx.x;
  if (n >= N) return;
  amax[n] = -__builtin_huge_valf();
  den[n]  = 0.f;
}

__global__ void gat_edge_max_kernel(const int* __restrict__ srcE, const int* __restrict__ dstE,
                                    int E, int ET, const float* __restrict__ aS,
                                    const float* __restrict__ aD, float* __restrict__ amax) {
  int e = blockIdx.x * blockDim.x + threadIdx.x;
  if (e >= ET) return;
  int s = e < E ? srcE[e] : e - E;
  int d = e < E ? dstE[e] : e - E;
  float a = aS[s] + aD[d];
  a = a > 0.f ? a : 0.2f * a;               // leaky_relu(0.2)
  atomicMaxF(&amax[d], a);
}

__global__ void gat_amax_fix_kernel(float* __restrict__ amax, int N) {
  int n = blockIdx.x * blockDim.x + threadIdx.x;
  if (n >= N) return;
  float v = amax[n];
  if (!__builtin_isfinite(v)) amax[n] = 0.f;
}

__global__ void gat_edge_sum_kernel(const int* __restrict__ srcE, const int* __restrict__ dstE,
                                    int E, int ET, const float* __restrict__ aS,
                                    const float* __restrict__ aD, const float* __restrict__ amax,
                                    float* __restrict__ exb, float* __restrict__ den) {
  int e = blockIdx.x * blockDim.x + threadIdx.x;
  if (e >= ET) return;
  int s = e < E ? srcE[e] : e - E;
  int d = e < E ? dstE[e] : e - E;
  float a = aS[s] + aD[d];
  a = a > 0.f ? a : 0.2f * a;
  float ex = expf(a - amax[d]);
  exb[e] = ex;
  atomicAdd(&den[d], ex);
}

__global__ void gat_bias_init_kernel(float* __restrict__ out, const float* __restrict__ bias,
                                     long long total, int mask) {
  long long i = (long long)blockIdx.x * blockDim.x + threadIdx.x;
  if (i < total) out[i] = bias[(int)(i & mask)];
}

// one thread per (edge, channel); gathers h[src] and atomically adds into out[dst]
__global__ void gat_aggregate_kernel(const int* __restrict__ srcE, const int* __restrict__ dstE,
                                     int E, const float* __restrict__ h,
                                     const float* __restrict__ exb, const float* __restrict__ den,
                                     float* __restrict__ out, int shf, int mask, long long total) {
  long long i = (long long)blockIdx.x * blockDim.x + threadIdx.x;
  if (i >= total) return;
  int c2 = (int)(i & mask);
  long long e = i >> shf;
  int s = e < E ? srcE[e] : (int)(e - E);
  int d = e < E ? dstE[e] : (int)(e - E);
  float coef = exb[e] / (den[d] + 1e-16f);
  atomicAdd(&out[((size_t)d << shf) + c2], h[((size_t)s << shf) + c2] * coef);
}

// ---------- residual ----------
__global__ void resid_skip_kernel(float* __restrict__ X, const float* __restrict__ Xs,
                                  const float* __restrict__ Y, long long n) {
  long long i = (long long)blockIdx.x * blockDim.x + threadIdx.x;
  if (i < n) X[i] = Xs[i] + Y[i];
}
__global__ void resid_inplace_kernel(float* __restrict__ X, const float* __restrict__ Y, long long n) {
  long long i = (long long)blockIdx.x * blockDim.x + threadIdx.x;
  if (i < n) X[i] += Y[i];
}

// =====================================================================
extern "C" void kernel_launch(void* const* d_in, const int* in_sizes, int n_in,
                              void* d_out, int out_size, void* d_ws, size_t ws_size,
                              hipStream_t stream) {
  (void)in_sizes; (void)n_in; (void)out_size; (void)ws_size;
  const int N = NNODES, E = NEDGES, ET = E + N;

  // ---- parse inputs in setup_inputs() dict order ----
  int cur = 0;
  auto F = [&]() { return (const float*)d_in[cur++]; };
  const float* x_in  = F();
  const float* fc1_W = F(); const float* fc1_b = F();
  struct Blk {
    const float *pre_w,*pre_b,*lin1_W,*lin1_b,*n1_w,*n1_b,
                *c1_W,*c1_as,*c1_ad,*c1_b,*n2_w,*n2_b,
                *c2_W,*c2_as,*c2_ad,*c2_b,*n3_w,*n3_b,*lin2_W,*lin2_b,*skip_W,*skip_b;
  } blk[8];
  const int in_cs[8]  = {256,128,128,128,128,128,128,128};
  const int out_cs[8] = {128,128,128,128,128,128,128, 64};
  for (int i = 0; i < 8; ++i) {
    Blk& p = blk[i];
    p.pre_w=F(); p.pre_b=F(); p.lin1_W=F(); p.lin1_b=F();
    p.n1_w=F();  p.n1_b=F();  p.c1_W=F();  p.c1_as=F(); p.c1_ad=F(); p.c1_b=F();
    p.n2_w=F();  p.n2_b=F();  p.c2_W=F();  p.c2_as=F(); p.c2_ad=F(); p.c2_b=F();
    p.n3_w=F();  p.n3_b=F();  p.lin2_W=F(); p.lin2_b=F();
    if (in_cs[i] != out_cs[i]) { p.skip_W = F(); p.skip_b = F(); }
    else                       { p.skip_W = nullptr; p.skip_b = nullptr; }
  }
  const float* gn_w = F(); const float* gn_b = F();
  const float* out_W = F(); const float* out_b = F();
  const int* ei   = (const int*)d_in[cur++];
  const int* srcE = ei;
  const int* dstE = ei + E;

  // ---- workspace carve ----
  char* wsp = (char*)d_ws;
  auto take = [&](size_t bytes) { void* p = wsp; wsp += (bytes + 255) & ~(size_t)255; return p; };
  float*          bufX  = (float*)take((size_t)N * 256 * 4);          // running activation
  unsigned short* stage = (unsigned short*)take((size_t)N * 256 * 2); // bf16 GEMM A staging
  float*          bufA  = (float*)take((size_t)N * 128 * 4);
  float*          bufB  = (float*)take((size_t)N * 128 * 4);
  float*          bufC  = (float*)take((size_t)N * 128 * 4);
  float*          aS    = (float*)take((size_t)N * 4);
  float*          aD    = (float*)take((size_t)N * 4);
  float*          amax  = (float*)take((size_t)N * 4);
  float*          den   = (float*)take((size_t)N * 4);
  float*          exb   = (float*)take((size_t)ET * 4);
  unsigned short* wbf   = (unsigned short*)take((size_t)65536 * 2);   // bf16 weight staging

  auto g1 = [](long long n, int bs) { return dim3((unsigned)((n + bs - 1) / bs)); };

  // WMMA GEMM with on-the-fly bf16 weight conversion; LDS-staged weight strips
  auto gemm = [&](const unsigned short* Abf, const float* Wf, int K, int Cout,
                  const float* bias, float* Cbuf) {
    cvt_bf16_kernel<<<g1((long long)Cout * K, 256), 256, 0, stream>>>(Wf, wbf, (long long)Cout * K);
    const unsigned gx = (unsigned)((N / 16 + 7) / 8);                 // 8 M-tiles per block
    if (Cout % 64 == 0) {
      wmma_gemm_kernel<4><<<dim3(gx, Cout / 64), 256, 0, stream>>>(Abf, wbf, bias, Cbuf, N, K, Cout);
    } else {
      wmma_gemm_kernel<2><<<dim3(gx, Cout / 32), 256, 0, stream>>>(Abf, wbf, bias, Cbuf, N, K, Cout);
    }
  };
  auto gnelu = [&](const float* xin, const float* wv, const float* bv, int C,
                   float* outf, unsigned short* outb) {
    gn_elu_kernel<<<g1((long long)N * (C / 8), 256), 256, 0, stream>>>(xin, wv, bv, outf, outb, N, C);
  };
  // GAT conv: input = stage (bf16, [N,half]); output (with bias) -> bufA
  auto gat = [&](int half, const float* Wc, const float* att_s, const float* att_d,
                 const float* bias) {
    gemm(stage, Wc, half, half, nullptr, bufB);                       // h = y @ W^T
    gat_scores_kernel<<<g1(N, 256), 256, 0, stream>>>(bufB, att_s, att_d, aS, aD, N, half);
    gat_init_kernel<<<g1(N, 256), 256, 0, stream>>>(amax, den, N);
    gat_edge_max_kernel<<<g1(ET, 256), 256, 0, stream>>>(srcE, dstE, E, ET, aS, aD, amax);
    gat_amax_fix_kernel<<<g1(N, 256), 256, 0, stream>>>(amax, N);
    gat_edge_sum_kernel<<<g1(ET, 256), 256, 0, stream>>>(srcE, dstE, E, ET, aS, aD, amax, exb, den);
    gat_bias_init_kernel<<<g1((long long)N * half, 256), 256, 0, stream>>>(bufA, bias,
                                                                           (long long)N * half, half - 1);
    const int shf = (half == 64) ? 6 : 5;
    const long long tot = (long long)ET << shf;
    gat_aggregate_kernel<<<g1(tot, 256), 256, 0, stream>>>(srcE, dstE, E, bufB, exb, den,
                                                           bufA, shf, half - 1, tot);
  };

  // ---- fc1: [N,23] -> [N,256] ----
  dense_naive_kernel<<<g1((long long)N * 256, 256), 256, 0, stream>>>(x_in, fc1_W, fc1_b, bufX, N, 23, 256);

  // ---- 8 blocks ----
  int in_c = 256;
  for (int bi = 0; bi < 8; ++bi) {
    const Blk& p = blk[bi];
    const int out_c = out_cs[bi], half = out_c / 2;
    gnelu(bufX, p.pre_w, p.pre_b, in_c, nullptr, stage);              // elu(gn(x)) -> bf16
    gemm(stage, p.lin1_W, in_c, half, p.lin1_b, bufA);                // lin1
    gnelu(bufA, p.n1_w, p.n1_b, half, nullptr, stage);
    gat(half, p.c1_W, p.c1_as, p.c1_ad, p.c1_b);                      // GAT 1 -> bufA
    gnelu(bufA, p.n2_w, p.n2_b, half, nullptr, stage);
    gat(half, p.c2_W, p.c2_as, p.c2_ad, p.c2_b);                      // GAT 2 -> bufA
    gnelu(bufA, p.n3_w, p.n3_b, half, nullptr, stage);
    gemm(stage, p.lin2_W, half, out_c, p.lin2_b, bufB);               // lin2 -> y in bufB
    if (p.skip_W) {
      cvt_bf16_kernel<<<g1((long long)N * in_c, 256), 256, 0, stream>>>(bufX, stage, (long long)N * in_c);
      gemm(stage, p.skip_W, in_c, out_c, p.skip_b, bufC);             // skip projection
      resid_skip_kernel<<<g1((long long)N * out_c, 256), 256, 0, stream>>>(bufX, bufC, bufB,
                                                                           (long long)N * out_c);
    } else {
      resid_inplace_kernel<<<g1((long long)N * out_c, 256), 256, 0, stream>>>(bufX, bufB,
                                                                              (long long)N * out_c);
    }
    in_c = out_c;
  }

  // ---- final GN+ELU then head [N,64] -> [N,3] ----
  gnelu(bufX, gn_w, gn_b, 64, bufA, nullptr);
  dense_naive_kernel<<<g1((long long)N * 3, 256), 256, 0, stream>>>(bufA, out_W, out_b,
                                                                    (float*)d_out, N, 64, 3);
}
